// MultiHeadAttention_41764261986665
// MI455X (gfx1250) — matrix operand order repaired and checked by reading, not compiled, not run on hardware
//
#include <hip/hip_runtime.h>
#include <hip/hip_bf16.h>

typedef __attribute__((ext_vector_type(16))) __bf16 v16bf;
typedef __attribute__((ext_vector_type(8)))  __bf16 v8bf;
typedef __attribute__((ext_vector_type(4)))  __bf16 v4bf;
typedef __attribute__((ext_vector_type(8)))  float  v8f;

#define NEG_INF_F (-4294967295.0f)        // -(2^32 - 1), matches reference mask constant
#define QK_SCALE  (0.17677669529663687f)  // 1/sqrt(32)

// ---------------------------------------------------------------------------
// Row-of-16 max reduction in 4 single VALU instructions using the DPP
// source modifier fused into v_max_num_f32 (ROW_XMASK keeps lanes 0-15 and
// 16-31 independent, matching the WMMA C-layout row placement).
// ---------------------------------------------------------------------------
__device__ __forceinline__ float row_max16(float x) {
  float a, b, c, d;
  asm("v_max_num_f32_dpp %0, %1, %1 row_xmask:1 row_mask:0xf bank_mask:0xf bound_ctrl:1"
      : "=v"(a) : "v"(x));
  asm("v_max_num_f32_dpp %0, %1, %1 row_xmask:2 row_mask:0xf bank_mask:0xf bound_ctrl:1"
      : "=v"(b) : "v"(a));
  asm("v_max_num_f32_dpp %0, %1, %1 row_xmask:4 row_mask:0xf bank_mask:0xf bound_ctrl:1"
      : "=v"(c) : "v"(b));
  asm("v_max_num_f32_dpp %0, %1, %1 row_xmask:8 row_mask:0xf bank_mask:0xf bound_ctrl:1"
      : "=v"(d) : "v"(c));
  return d;
}
__device__ __forceinline__ float row_bcast0(float x) {
  int t = __builtin_amdgcn_update_dpp(0, __float_as_int(x), 0x150 /*ROW_SHARE:0*/,
                                      0xf, 0xf, true);
  return __int_as_float(t);
}

// ---------------------------------------------------------------------------
// C = A[M x 256] * W[256 x 256] + bias, tiled 64x64 per block (4 waves).
// A is fp32 or bf16; output is bf16 (scaled) or fp32.
// All MACs via v_wmma_f32_16x16x32_bf16. Staging moves 16B per thread.
// ---------------------------------------------------------------------------
template <bool A_BF16, bool OUT_F32>
__global__ __launch_bounds__(128)
void gemm256_bias_kernel(const void* __restrict__ Av,
                         const float* __restrict__ W,
                         const float* __restrict__ bias,
                         void* __restrict__ outv,
                         int M, float outScale) {
  __shared__ __bf16 As[64][32];
  __shared__ __bf16 Bs[32][64];

  const int tid  = threadIdx.x;
  const int wave = tid >> 5;
  const int lane = tid & 31;
  const int half = lane >> 4;
  const int l15  = lane & 15;
  const int m0   = blockIdx.x * 64;
  const int n0   = blockIdx.y * 64;

  const v8f vzero = {0.f, 0.f, 0.f, 0.f, 0.f, 0.f, 0.f, 0.f};
  v8f acc[4];
#pragma unroll
  for (int t = 0; t < 4; ++t) acc[t] = vzero;

  for (int kc = 0; kc < 256; kc += 32) {
    // ---- Stage A tile (64x32) as bf16, vectorized ----
    if constexpr (A_BF16) {
#pragma unroll
      for (int i = 0; i < 2; ++i) {
        int e8 = tid + i * 128;            // 256 x v8bf chunks
        int r = e8 >> 2, c = (e8 & 3) * 8;
        *(v8bf*)&As[r][c] =
            *(const v8bf*)((const __bf16*)Av + (size_t)(m0 + r) * 256 + kc + c);
      }
    } else {
#pragma unroll
      for (int i = 0; i < 4; ++i) {
        int e4 = tid + i * 128;            // 512 x float4 chunks
        int r = e4 >> 3, c = (e4 & 7) * 4;
        float4 f = *(const float4*)((const float*)Av + (size_t)(m0 + r) * 256 + kc + c);
        v4bf h;
        h[0] = (__bf16)f.x; h[1] = (__bf16)f.y; h[2] = (__bf16)f.z; h[3] = (__bf16)f.w;
        *(v4bf*)&As[r][c] = h;
      }
    }
    // ---- Stage W tile (32x64) as bf16, vectorized ----
#pragma unroll
    for (int i = 0; i < 4; ++i) {
      int e4 = tid + i * 128;
      int r = e4 >> 4, c = (e4 & 15) * 4;
      float4 f = *(const float4*)(W + (size_t)(kc + r) * 256 + n0 + c);
      v4bf h;
      h[0] = (__bf16)f.x; h[1] = (__bf16)f.y; h[2] = (__bf16)f.z; h[3] = (__bf16)f.w;
      *(v4bf*)&Bs[r][c] = h;
    }
    __syncthreads();

    // A fragment: rows wave*16 .. wave*16+15, K=0..31 (k = 16*half + element, same order as B)
    v16bf a;
#pragma unroll
    for (int v = 0; v < 8; ++v) {
      int k0 = half * 16 + 2 * v;
      a[2 * v]     = As[wave * 16 + l15][k0];
      a[2 * v + 1] = As[wave * 16 + l15][k0 + 1];
    }
#pragma unroll
    for (int t = 0; t < 4; ++t) {
      v16bf b;
#pragma unroll
      for (int v = 0; v < 8; ++v) {
        int k0 = half * 16 + 2 * v;
        b[2 * v]     = Bs[k0][t * 16 + l15];
        b[2 * v + 1] = Bs[k0 + 1][t * 16 + l15];
      }
      acc[t] = __builtin_amdgcn_wmma_f32_16x16x32_bf16(
          false, a, false, b, (short)0, acc[t], false, false);
    }
    __syncthreads();
  }

  // Epilogue: C layout row = v + 8*half, col = lane&15
#pragma unroll
  for (int t = 0; t < 4; ++t) {
    int col = n0 + t * 16 + l15;
    float bv = bias[col];
#pragma unroll
    for (int v = 0; v < 8; ++v) {
      int row = m0 + wave * 16 + v + 8 * half;
      float val = acc[t][v] + bv;
      size_t g = (size_t)row * 256 + col;
      if constexpr (OUT_F32) {
        ((float*)outv)[g] = val;
      } else {
        ((__bf16*)outv)[g] = (__bf16)(val * outScale);
      }
    }
  }
}

// ---------------------------------------------------------------------------
// Flash attention: one block = 64 query rows of one (batch, head).
// 4 waves, each owns 16 query rows. K/V tiles of 32 keys staged in LDS.
// Per key chunk: 2 WMMAs for Q*K^T, 2 WMMAs for P*V, and 1 WMMA that
// accumulates the softmax row-sum (P x ones-column) on the matrix pipe.
// Row max via single-instruction DPP16 xor-reduction steps (VALU only).
// Q is pre-scaled by 1/sqrt(depth) at projection time.
// ---------------------------------------------------------------------------
__global__ __launch_bounds__(128)
void flash_attn_kernel(const __bf16* __restrict__ Q,
                       const __bf16* __restrict__ K,
                       const __bf16* __restrict__ V,
                       const int* __restrict__ key_mask,
                       __bf16* __restrict__ O, int L) {
  __shared__ __bf16 Kt[32][32];        // [key][depth]
  __shared__ __bf16 Vt[32][32];        // [key][depth]
  __shared__ float  mAdd[32];          // additive key mask for this chunk
  __shared__ __bf16 Pb[4][16][32];     // per-wave P staging (C-layout -> A-layout)

  const int tid  = threadIdx.x;
  const int wave = tid >> 5;
  const int lane = tid & 31;
  const int half = lane >> 4;
  const int l15  = lane & 15;

  const int bh   = blockIdx.y;
  const int b    = bh >> 3;
  const int head = bh & 7;
  const size_t rowOff = (size_t)b * L;
  const int hd   = head * 32;
  const int q0   = blockIdx.x * 64 + wave * 16;

  // Q fragment (held in registers across the whole key loop)
  v16bf qf;
  {
    const __bf16* qp = Q + (rowOff + q0 + l15) * 256 + hd;
#pragma unroll
    for (int v = 0; v < 8; ++v) {
      int k0 = half * 16 + 2 * v;
      qf[2 * v]     = qp[k0];
      qf[2 * v + 1] = qp[k0 + 1];
    }
  }

  // Constant B fragment: ones in column 0 -> row-sum WMMA
  v16bf onesf;
  {
    __bf16 one = (__bf16)1.0f, zer = (__bf16)0.0f;
#pragma unroll
    for (int v = 0; v < 16; ++v) onesf[v] = (l15 == 0) ? one : zer;
  }

  const v8f vzero = {0.f, 0.f, 0.f, 0.f, 0.f, 0.f, 0.f, 0.f};
  v8f o0 = vzero, o1 = vzero;           // 16 x 32 (depth) fp32 accum
  v8f lacc = vzero;                     // row sums live in lanes with l15==0
  float m[8];
#pragma unroll
  for (int v = 0; v < 8; ++v) m[v] = -3.0e38f;

  for (int kc = 0; kc < L; kc += 32) {
    // Stage K and V chunks (32 keys x 32 depth), 16B per thread per matrix
    {
      int r = tid >> 2, c = (tid & 3) * 8;
      size_t g = (rowOff + kc + r) * 256 + hd + c;
      *(v8bf*)&Kt[r][c] = *(const v8bf*)(K + g);
      *(v8bf*)&Vt[r][c] = *(const v8bf*)(V + g);
    }
    if (tid < 32)
      mAdd[tid] = (key_mask[rowOff + kc + tid] != 0) ? 0.f : NEG_INF_F;
    __syncthreads();

    // Scores: S(16x32) = Q(16x32depth) * K^T  -> two 16x16 WMMAs
    v16bf bf0, bf1;
#pragma unroll
    for (int v = 0; v < 8; ++v) {
      int k0 = half * 16 + 2 * v;              // depth index
      bf0[2 * v]     = Kt[l15][k0];            // keys 0..15
      bf0[2 * v + 1] = Kt[l15][k0 + 1];
      bf1[2 * v]     = Kt[16 + l15][k0];       // keys 16..31
      bf1[2 * v + 1] = Kt[16 + l15][k0 + 1];
    }
    v8f s0 = __builtin_amdgcn_wmma_f32_16x16x32_bf16(
        false, qf, false, bf0, (short)0, vzero, false, false);
    v8f s1 = __builtin_amdgcn_wmma_f32_16x16x32_bf16(
        false, qf, false, bf1, (short)0, vzero, false, false);

    // Online softmax, rows v + 8*half, cols = key lane
    float add0 = mAdd[l15];
    float add1 = mAdd[16 + l15];
#pragma unroll
    for (int v = 0; v < 8; ++v) {
      float x0 = s0[v] + add0;
      float x1 = s1[v] + add1;
      float mx = row_max16(fmaxf(x0, x1));     // 4x v_max_num_f32_dpp
      float mn   = fmaxf(m[v], mx);
      float corr = __expf(m[v] - mn);
      m[v] = mn;
      float p0 = __expf(x0 - mn);
      float p1 = __expf(x1 - mn);
      o0[v]   = o0[v] * corr;
      o1[v]   = o1[v] * corr;
      lacc[v] = lacc[v] * corr;
      // Stage P (C-layout) into LDS for A-layout reload
      Pb[wave][v + 8 * half][l15]      = (__bf16)p0;
      Pb[wave][v + 8 * half][16 + l15] = (__bf16)p1;
    }
    __syncthreads();

    // O += P(16x32keys) * V ; lacc += P * ones_col0 (row sums on matrix pipe)
    v16bf pf, vb0, vb1;
#pragma unroll
    for (int v = 0; v < 8; ++v) {
      int k0 = half * 16 + 2 * v;
      pf[2 * v]     = Pb[wave][l15][k0];
      pf[2 * v + 1] = Pb[wave][l15][k0 + 1];
    }
#pragma unroll
    for (int v = 0; v < 8; ++v) {
      int k0 = half * 16 + 2 * v;              // key index
      vb0[2 * v]     = Vt[k0][l15];            // depth 0..15
      vb0[2 * v + 1] = Vt[k0 + 1][l15];
      vb1[2 * v]     = Vt[k0][16 + l15];       // depth 16..31
      vb1[2 * v + 1] = Vt[k0 + 1][16 + l15];
    }
    o0 = __builtin_amdgcn_wmma_f32_16x16x32_bf16(
        false, pf, false, vb0, (short)0, o0, false, false);
    o1 = __builtin_amdgcn_wmma_f32_16x16x32_bf16(
        false, pf, false, vb1, (short)0, o1, false, false);
    lacc = __builtin_amdgcn_wmma_f32_16x16x32_bf16(
        false, pf, false, onesf, (short)0, lacc, false, false);
    __syncthreads();
  }

  // Normalize and write attention output (bf16, [B*L, 256] with head offset)
#pragma unroll
  for (int v = 0; v < 8; ++v) {
    float s = row_bcast0(lacc[v]);             // broadcast col-0 sum to the row
    float inv = 1.0f / s;
    int row = q0 + v + 8 * half;
    size_t g = (rowOff + row) * 256 + hd;
    O[g + l15]      = (__bf16)(o0[v] * inv);
    O[g + 16 + l15] = (__bf16)(o1[v] * inv);
  }
}

// ---------------------------------------------------------------------------
// Launch: 3 projection GEMMs -> flash attention -> output GEMM.
// Workspace layout (bf16): Q | K | V | attn_out, each M*256 elements (8 MB).
// ---------------------------------------------------------------------------
extern "C" void kernel_launch(void* const* d_in, const int* in_sizes, int n_in,
                              void* d_out, int out_size, void* d_ws, size_t ws_size,
                              hipStream_t stream) {
  const float* queries  = (const float*)d_in[0];
  const float* keys     = (const float*)d_in[1];
  const float* values   = (const float*)d_in[2];
  const int*   key_mask = (const int*)d_in[3];
  const float* Wq = (const float*)d_in[4];
  const float* bq = (const float*)d_in[5];
  const float* Wk = (const float*)d_in[6];
  const float* bk = (const float*)d_in[7];
  const float* Wv = (const float*)d_in[8];
  const float* bv = (const float*)d_in[9];
  const float* Wo = (const float*)d_in[10];
  const float* bo = (const float*)d_in[11];
  float* out = (float*)d_out;

  const int D = 256;
  const int M = in_sizes[0] / D;   // B * L
  const int L = 2048;
  const int B = M / L;

  __bf16* Qbf = (__bf16*)d_ws;
  __bf16* Kbf = Qbf + (size_t)M * D;
  __bf16* Vbf = Kbf + (size_t)M * D;
  __bf16* Abf = Vbf + (size_t)M * D;

  dim3 blk(128);
  dim3 gGemm(M / 64, D / 64);

  // Projections (Q gets 1/sqrt(depth) folded in)
  gemm256_bias_kernel<false, false><<<gGemm, blk, 0, stream>>>(
      queries, Wq, bq, Qbf, M, QK_SCALE);
  gemm256_bias_kernel<false, false><<<gGemm, blk, 0, stream>>>(
      keys, Wk, bk, Kbf, M, 1.0f);
  gemm256_bias_kernel<false, false><<<gGemm, blk, 0, stream>>>(
      values, Wv, bv, Vbf, M, 1.0f);

  // Attention
  dim3 gAttn(L / 64, B * 8);
  flash_attn_kernel<<<gAttn, blk, 0, stream>>>(Qbf, Kbf, Vbf, key_mask, Abf, L);

  // Output projection (fp32 out)
  gemm256_bias_kernel<true, true><<<gGemm, blk, 0, stream>>>(
      Abf, Wo, bo, out, M, 1.0f);
}
